// GraphEncoder_30528627540168
// MI455X (gfx1250) — matrix operand order repaired and checked by reading, not compiled, not run on hardware
//
#include <hip/hip_runtime.h>
#include <cstddef>

// ---------------------------------------------------------------------------
// MI455X (gfx1250) implementation of the gated graph encoder.
//
// Dominant cost: uzM einsum == per-(b,i) GEMM (2048x32)x(32x64) fused with
// floor(sigmoid()) gate and k-reduction (17.2 GFLOP total). Implemented with
// v_wmma_f32_16x16x32_bf16 using bf16 hi/lo split of u_z for ~fp32 accuracy.
// Weights (67MB) are L2-resident; uzM is never materialized.
// Gate/z sigmoids use hardware v_rcp_f32 (1 TRANS op) instead of the IEEE
// division macro — the epilogue VALU stream must co-issue with the WMMAs.
// ---------------------------------------------------------------------------

typedef __attribute__((ext_vector_type(16))) __bf16 v16bf;
typedef __attribute__((ext_vector_type(8)))  __bf16 v8bf;
typedef __attribute__((ext_vector_type(8)))  float  v8f;

union BFU { __bf16 h; unsigned short u; };

static __device__ __forceinline__ unsigned short f2bf(float x) {
    BFU t; t.h = (__bf16)x; return t.u;
}
static __device__ __forceinline__ float bf2f(unsigned short b) {
    BFU t; t.u = b; return (float)t.h;
}
// fast sigmoid: v_exp_f32 + v_add + v_rcp_f32 (no division macro)
static __device__ __forceinline__ float sigm_fast(float x) {
    return __builtin_amdgcn_rcpf(1.f + __expf(-x));
}

// ---- split u_z (f32, 64*64*32*32 elems) into bf16 hi/lo --------------------
__global__ void __launch_bounds__(256)
k_split_uz(const float* __restrict__ uz,
           unsigned short* __restrict__ uzh, unsigned short* __restrict__ uzl) {
    const size_t base = ((size_t)blockIdx.x * 256 + threadIdx.x) * 4;
    const float4 v = *(const float4*)(uz + base);
    float xs[4] = {v.x, v.y, v.z, v.w};
    unsigned long long H = 0, L = 0;
#pragma unroll
    for (int e = 0; e < 4; ++e) {
        unsigned short h = f2bf(xs[e]);
        unsigned short l = f2bf(xs[e] - bf2f(h));
        H |= ((unsigned long long)h) << (16 * e);
        L |= ((unsigned long long)l) << (16 * e);
    }
    *(unsigned long long*)(uzh + base) = H;
    *(unsigned long long*)(uzl + base) = L;
}

// ---- wzx/wcx precompute: wzx[b,i,j,f] = dot(w_z[i,j,f,:], x[b,i,:]) --------
__global__ void __launch_bounds__(256)
k_wx(const float* __restrict__ x, const float* __restrict__ wz,
     const float* __restrict__ wc, float* __restrict__ wzx, float* __restrict__ wcx) {
    __shared__ float xL[32];
    const int tid = threadIdx.x;
    const int blk = blockIdx.x;           // b*64 + i
    const int i = blk & 63;
    if (tid < 32) xL[tid] = x[(size_t)blk * 32 + tid];
    __syncthreads();
    for (int s = tid; s < 2048; s += 256) {
        const float* pz = wz + ((size_t)i * 2048 + s) * 32;
        const float* pc = wc + ((size_t)i * 2048 + s) * 32;
        float az = 0.f, ac = 0.f;
#pragma unroll
        for (int g = 0; g < 32; ++g) { az += pz[g] * xL[g]; ac += pc[g] * xL[g]; }
        wzx[(size_t)blk * 2048 + s] = az;
        wcx[(size_t)blk * 2048 + s] = ac;
    }
}

// ---- one message-passing iteration for one (b,i): WMMA GEMM + fused gate ---
__global__ void __launch_bounds__(256)
k_iter(const float* __restrict__ Adj, const float* __restrict__ wzx,
       const float* __restrict__ wcx, const float* __restrict__ bz,
       const float* __restrict__ bc, const float* __restrict__ uc,
       const unsigned short* __restrict__ uzh, const unsigned short* __restrict__ uzl,
       const float* __restrict__ mprev, float* __restrict__ mfout, int first) {
    __shared__ float MiF[64 * 33];                        // padded: conflict-free column reads
    __shared__ __align__(64) unsigned short MiH[64 * 32]; // bf16 hi of M[i]
    __shared__ __align__(64) unsigned short MiLo[64 * 32];// bf16 lo of M[i]
    __shared__ float wzxbL[2048];                         // wzx + b_z
    __shared__ float gsumL[2048];
    __shared__ float zpreL[2048];
    __shared__ float ArowL[64];
    __shared__ float aggL[32];

    const int tid = threadIdx.x;
    const int blk = blockIdx.x;          // b*64 + i
    const int b = blk >> 6;
    const int i = blk & 63;
    const size_t bi = (size_t)blk;

    // phase 0a: adjacency row + (wzx + b_z)
    if (tid < 64) ArowL[tid] = Adj[bi * 64 + tid];
    for (int s = tid; s < 2048; s += 256)
        wzxbL[s] = wzx[bi * 2048 + s] + bz[s & 31];
    __syncthreads();

    // phase 0b: M[i,k,f] = A[i,k] * m_prev[max(i,k), min(i,k), f]  (0 at t=0)
    for (int s = tid; s < 2048; s += 256) {
        const int k = s >> 5, f = s & 31;
        float m = 0.f;
        if (!first) {
            const int hi = i > k ? i : k, lo = i > k ? k : i;
            m = ArowL[k] * mprev[(((size_t)b * 64 + hi) * 64 + lo) * 32 + f];
        }
        MiF[k * 33 + f] = m;
        const unsigned short h = f2bf(m);
        MiH[s] = h;
        MiLo[s] = f2bf(m - bf2f(h));
    }
    __syncthreads();
    if (tid < 32) {
        float a = 0.f;
#pragma unroll 8
        for (int k = 0; k < 64; ++k) a += ArowL[k] * MiF[k * 33 + tid];
        aggL[tid] = a;
    }
    __syncthreads();

    // phase 1: G[jf,k] = u_z[i] @ M[i]^T via bf16-split WMMA, fused epilogue
    const int w = tid >> 5;
    const int lane = tid & 31;
    const int ln = lane & 15;
    const int lh = lane >> 4;
    const size_t uzbase = (size_t)i * 2048 * 32;

    for (int it = 0; it < 16; ++it) {
        const int tM = w + (it << 3);        // 128 M-tiles over 8 waves
        const int j = tM >> 1;               // j is constant within an M-tile
        const float aj = ArowL[j];

        // A-tile (16x32 bf16): lane r -> K0-7,K16-23 ; lane r+16 -> K8-15,K24-31
        const size_t rowoff = uzbase + (size_t)(tM * 16 + ln) * 32;
        const v8bf h0 = *(const v8bf*)((const __bf16*)uzh + rowoff + lh * 8);
        const v8bf h1 = *(const v8bf*)((const __bf16*)uzh + rowoff + 16 + lh * 8);
        const v8bf l0 = *(const v8bf*)((const __bf16*)uzl + rowoff + lh * 8);
        const v8bf l1 = *(const v8bf*)((const __bf16*)uzl + rowoff + 16 + lh * 8);
        const v16bf Ah = __builtin_shufflevector(h0, h1, 0,1,2,3,4,5,6,7,8,9,10,11,12,13,14,15);
        const v16bf Al = __builtin_shufflevector(l0, l1, 0,1,2,3,4,5,6,7,8,9,10,11,12,13,14,15);

        float gs[8], zsum[8], gjj[8];
#pragma unroll
        for (int v = 0; v < 8; ++v) { gs[v] = 0.f; zsum[v] = 0.f; gjj[v] = 0.f; }

#pragma unroll
        for (int nt = 0; nt < 4; ++nt) {
            // B-tile (32x16 bf16): lane n -> K0-7,K16-23 ; lane n+16 -> K8-15,K24-31
            const __bf16* pBh = (const __bf16*)MiH + (nt * 16 + ln) * 32 + lh * 8;
            const __bf16* pBl = (const __bf16*)MiLo + (nt * 16 + ln) * 32 + lh * 8;
            const v16bf Bh = __builtin_shufflevector(*(const v8bf*)pBh, *(const v8bf*)(pBh + 16), 0,1,2,3,4,5,6,7,8,9,10,11,12,13,14,15);
            const v16bf Bl = __builtin_shufflevector(*(const v8bf*)pBl, *(const v8bf*)(pBl + 16), 0,1,2,3,4,5,6,7,8,9,10,11,12,13,14,15);
            v8f c = {0.f, 0.f, 0.f, 0.f, 0.f, 0.f, 0.f, 0.f};
            c = __builtin_amdgcn_wmma_f32_16x16x32_bf16(false, Ah, false, Bh, (short)0, c, false, false);
            c = __builtin_amdgcn_wmma_f32_16x16x32_bf16(false, Ah, false, Bl, (short)0, c, false, false);
            c = __builtin_amdgcn_wmma_f32_16x16x32_bf16(false, Al, false, Bh, (short)0, c, false, false);

            const int k = nt * 16 + ln;
            const float ak = ArowL[k];
            const float km = (k == j) ? 0.f : ak;     // A[i,k] * (k != j)
            const float kj = (k == j) ? 1.f : 0.f;
#pragma unroll
            for (int v = 0; v < 8; ++v) {
                const int jf = (tM << 4) + v + (lh << 3);   // C/D: lane half -> M=v / M=v+8
                const int f = jf & 31;
                const float G = c[v];
                const float gate = floorf(sigm_fast(wzxbL[jf] + G));
                gs[v]   += gate * km * MiF[k * 33 + f];     // gated_sum partial
                zsum[v] += ak * G;                          // u_z @ agg  partial
                gjj[v]  += kj * G;                          // G[jf, k=j]
            }
        }
#pragma unroll
        for (int v = 0; v < 8; ++v) {  // reduce over the 16 columns in each half
            float a = gs[v], z2 = zsum[v], g2 = gjj[v];
#pragma unroll
            for (int m = 8; m >= 1; m >>= 1) {
                a  += __shfl_xor(a,  m, 32);
                z2 += __shfl_xor(z2, m, 32);
                g2 += __shfl_xor(g2, m, 32);
            }
            if (ln == 0) {
                const int jf = (tM << 4) + v + (lh << 3);
                gsumL[jf] = a;
                // u_z@m_other = u_z@agg - A[i,j]*(u_z@M[i,j]) recovered from GEMM
                zpreL[jf] = wzxbL[jf] + z2 - aj * g2;
            }
        }
    }
    __syncthreads();

    // phase 2: r_msg = u_c@gated_sum ; cur = tanh(wcx + u_c@r_msg + b_c); blend
    const int f = lane;
    for (int q = 0; q < 8; ++q) {
        const int j = w + (q << 3);
        const float* Ur = uc + (((size_t)i * 64 + j) * 32 + f) * 32;
        float urow[32];
#pragma unroll
        for (int g = 0; g < 32; ++g) urow[g] = Ur[g];
        float rm = 0.f;
#pragma unroll
        for (int g = 0; g < 32; ++g) rm += urow[g] * gsumL[(j << 5) + g];
        float acc = 0.f;
#pragma unroll
        for (int g = 0; g < 32; ++g) acc += urow[g] * __shfl(rm, g, 32);
        const float cur = tanhf(wcx[bi * 2048 + (j << 5) + f] + acc + bc[f]);
        const float zv = sigm_fast(zpreL[(j << 5) + f]);
        const float mo = aggL[f] - ArowL[j] * MiF[j * 33 + f];
        mfout[bi * 2048 + (j << 5) + f] = (1.f - zv) * mo + zv * cur;
    }
}

// ---- final node encoding: enc = relu(u_node@x + u_msg@msg_sum) -------------
__global__ void k_enc(const float* __restrict__ x, const float* __restrict__ Adj,
                      const float* __restrict__ mf, const float* __restrict__ un,
                      const float* __restrict__ um, float* __restrict__ enc) {
    const int blk = blockIdx.x;          // b*64 + i
    const int b = blk >> 6;
    const int i = blk & 63;
    const int f = threadIdx.x;           // 0..31, one wave
    float ms = 0.f;
    for (int j = 0; j < 64; ++j) {
        const int hi = i > j ? i : j, lo = i > j ? j : i;
        ms += Adj[(size_t)blk * 64 + j] * mf[(((size_t)b * 64 + hi) * 64 + lo) * 32 + f];
    }
    const float xv = x[(size_t)blk * 32 + f];
    const float* pn = un + ((size_t)i * 32 + f) * 32;
    const float* pm = um + ((size_t)i * 32 + f) * 32;
    float acc = 0.f;
#pragma unroll
    for (int g = 0; g < 32; ++g)
        acc += pn[g] * __shfl(xv, g, 32) + pm[g] * __shfl(ms, g, 32);
    enc[(size_t)b * 2048 + i * 32 + f] = fmaxf(acc, 0.f);
}

// ---- output linear + sigmoid (precise division: feeds final output) -------
__global__ void k_lin(const float* __restrict__ enc, const float* __restrict__ lw,
                      const float* __restrict__ lb, float* __restrict__ out) {
    const int b = blockIdx.x;
    const int o = threadIdx.x;           // 0..127
    const float* wr = lw + (size_t)o * 2048;
    const float* er = enc + (size_t)b * 2048;
    float acc = lb[o];
    for (int s = 0; s < 2048; ++s) acc += wr[s] * er[s];
    out[(size_t)b * 128 + o] = 1.f / (1.f + __expf(-acc));
}

extern "C" void kernel_launch(void* const* d_in, const int* in_sizes, int n_in,
                              void* d_out, int out_size, void* d_ws, size_t ws_size,
                              hipStream_t stream) {
    (void)in_sizes; (void)n_in; (void)out_size; (void)ws_size;
    const float* x   = (const float*)d_in[0];
    const float* Adj = (const float*)d_in[1];
    const float* wz  = (const float*)d_in[2];
    const float* wc  = (const float*)d_in[3];
    const float* uz  = (const float*)d_in[4];
    const float* uc  = (const float*)d_in[5];
    const float* bz  = (const float*)d_in[6];
    const float* bc  = (const float*)d_in[7];
    const float* un  = (const float*)d_in[8];
    const float* um  = (const float*)d_in[9];
    const float* lw  = (const float*)d_in[10];
    const float* lb  = (const float*)d_in[11];
    float* out = (float*)d_out;

    // workspace carve-up (~32 MB)
    char* p = (char*)d_ws;
    unsigned short* uzh = (unsigned short*)p; p += (size_t)4194304 * 2;
    unsigned short* uzl = (unsigned short*)p; p += (size_t)4194304 * 2;
    float* wzx = (float*)p; p += (size_t)1048576 * 4;
    float* wcx = (float*)p; p += (size_t)1048576 * 4;
    float* mfA = (float*)p; p += (size_t)1048576 * 4;
    float* mfB = (float*)p; p += (size_t)1048576 * 4;
    float* enc = (float*)p; p += (size_t)16384 * 4;

    k_split_uz<<<4096, 256, 0, stream>>>(uz, uzh, uzl);
    k_wx<<<512, 256, 0, stream>>>(x, wz, wc, wzx, wcx);

    // T = 4 message-passing iterations; launch boundary = global sync for
    // the cross-i symmetrization M <- A * m_full[hi,lo]; ping-pong buffers.
    k_iter<<<512, 256, 0, stream>>>(Adj, wzx, wcx, bz, bc, uc, uzh, uzl, mfA, mfA, 1);
    k_iter<<<512, 256, 0, stream>>>(Adj, wzx, wcx, bz, bc, uc, uzh, uzl, mfA, mfB, 0);
    k_iter<<<512, 256, 0, stream>>>(Adj, wzx, wcx, bz, bc, uc, uzh, uzl, mfB, mfA, 0);
    k_iter<<<512, 256, 0, stream>>>(Adj, wzx, wcx, bz, bc, uc, uzh, uzl, mfA, mfB, 0);

    k_enc<<<512, 32, 0, stream>>>(x, Adj, mfB, un, um, enc);
    k_lin<<<8, 128, 0, stream>>>(enc, lw, lb, out);
}
